// actor_34540126995072
// MI455X (gfx1250) — matrix operand order repaired and checked by reading, not compiled, not run on hardware
//
#include <hip/hip_runtime.h>
#include <hip/hip_bf16.h>

// ---------------------------------------------------------------------------
// GAT-TSP actor head for MI455X (gfx1250, wave32, WMMA + TDM).
//   * GEMMs on v_wmma_f32_16x16x32_f16 (f16 in, f32 accumulate).
//   * B (weights) pre-packed into per-lane fragment order -> b128 loads.
//   * A tiles DMA'd to LDS with tensor_load_to_lds (TENSORcnt), read back
//     as ds_load_b128 fragments.
// Working set ~40MB -> L2 resident -> matrix-pipe bound.
// ---------------------------------------------------------------------------

#define NN     6400      // nodes
#define GG     128       // graphs
#define CC     50        // cities per graph
#define HH     128       // hidden
#define DD     256       // GAT dim (2*H)
#define HEADS  3
#define K1     256       // GEMM1 reduction (2H)
#define NC1    768       // GEMM1 cols (HEADS*D)
#define K2     512       // GEMM2 reduction (2D)
#define NC2    128       // GEMM2 cols (H)
#define NEG_INF 1.0e6f

typedef __attribute__((ext_vector_type(16))) _Float16     v16h;
typedef __attribute__((ext_vector_type(8)))  _Float16     h8;
typedef __attribute__((ext_vector_type(8)))  float        v8f;
typedef __attribute__((ext_vector_type(4)))  unsigned int u32x4;
typedef __attribute__((ext_vector_type(8)))  int          i32x8;
typedef __attribute__((ext_vector_type(4)))  int          i32x4;

#if defined(__has_builtin)
#if __has_builtin(__builtin_amdgcn_tensor_load_to_lds)
#define HAVE_TDM 1
#endif
#endif

// ---- TDM: DMA a contiguous [tile_rows x row_len] f16 tile into LDS ---------
// D# layout per CDNA5 ISA 8.3/8.4: 2D tensor, data_size=2B.
#ifdef HAVE_TDM
__device__ __forceinline__ void tdm_load_tile_f16(unsigned int lds_off,
                                                  const _Float16* gptr,
                                                  unsigned int row_len,
                                                  unsigned int tensor_rows,
                                                  unsigned int tile_rows) {
    unsigned long long ga = (unsigned long long)(uintptr_t)gptr;
    u32x4 g0;
    g0[0] = 1u;                                   // count=1 (valid user D#)
    g0[1] = lds_off;                              // lds_addr (bytes)
    g0[2] = (unsigned int)ga;                     // global_addr[31:0]
    g0[3] = (unsigned int)(ga >> 32) | (2u << 30);// global_addr[56:32] | type=2
    i32x8 g1;
    g1[0] = 1 << 16;                              // wg_mask=0 | data_size=2B
    g1[1] = (int)(row_len << 16);                 // tensor_dim0[15:0] @ bits63:48
    g1[2] = (int)(tensor_rows << 16);             // dim0[31:16]=0 | tensor_dim1[15:0]
    g1[3] = (int)(row_len << 16);                 // dim1[31:16]=0 | tile_dim0
    g1[4] = (int)tile_rows;                       // tile_dim1 | tile_dim2=0
    g1[5] = (int)row_len;                         // tensor_dim0_stride[31:0]
    g1[6] = 0;                                    // stride0[47:32] | stride1[15:0]
    g1[7] = 0;                                    // stride1[47:16]
    i32x4 z4 = {0, 0, 0, 0};
#if __clang_major__ >= 23
    i32x8 z8 = {0, 0, 0, 0, 0, 0, 0, 0};
    __builtin_amdgcn_tensor_load_to_lds(g0, g1, z4, z4, z8, 0);
#else
    __builtin_amdgcn_tensor_load_to_lds(g0, g1, z4, z4, 0);
#endif
    __builtin_amdgcn_s_wait_tensorcnt(0);
}
#endif

// ---- WMMA fragment loaders (layouts per CDNA5 ISA 7.12.2) ------------------

// A 16x32 f16 fragment from LDS (row-major tile, leading dim ldk).
// lanes 0-15: M=lane,  K = k0+{0..7, 16..23}
// lanes16-31: M=ln-16, K = k0+{8..15, 24..31}
__device__ __forceinline__ v16h lds_a_frag(const _Float16* sh, int ldk,
                                           int row, int k0, int lane) {
    const _Float16* p = sh + row * ldk + k0 + ((lane & 16) ? 8 : 0);
    h8 lo = *(const h8*)(p);       // ds_load_b128
    h8 hi = *(const h8*)(p + 16);  // ds_load_b128
    v16h a;
#pragma unroll
    for (int j = 0; j < 8; ++j) { a[j] = lo[j]; a[8 + j] = hi[j]; }
    return a;
}

// B fragment from pre-packed layout: [ktile][ntile][lane][16 halves].
__device__ __forceinline__ v16h packed_b_frag(const _Float16* Bp, int ntiles_n,
                                              int k0, int ntile, int lane) {
    size_t tile = (size_t)(k0 >> 5) * ntiles_n + ntile;
    return *(const v16h*)(Bp + (tile << 9) + (lane << 4));  // 32B contiguous
}

// ---- pack weights f32[K x NC] -> f16 fragment order ------------------------
__global__ void __launch_bounds__(256)
pack_b_kernel(const float* __restrict__ src, _Float16* __restrict__ dst,
              int K, int NC) {
    int idx = blockIdx.x * 256 + threadIdx.x;
    if (idx >= K * NC) return;
    int j    = idx & 15;             // half within lane
    int lane = (idx >> 4) & 31;
    int tile = idx >> 9;
    int ntiles_n = NC >> 4;
    int kt = tile / ntiles_n, nt = tile % ntiles_n;
    int col = nt * 16 + (lane & 15);
    int row = kt * 32 + ((lane & 16) ? 8 : 0) + (j & 7) + ((j & 8) ? 16 : 0);
    dst[idx] = (_Float16)src[(size_t)row * NC + col];
}

// ---- prep: x0 = x@w0 + b0, x_in = concat(x0[prev], x0) in f16 --------------
__global__ void __launch_bounds__(256)
prep_xin_kernel(const float* __restrict__ x, const int* __restrict__ prev,
                const float* __restrict__ w0, const float* __restrict__ b0,
                _Float16* __restrict__ xin) {
    int idx = blockIdx.x * 256 + threadIdx.x;          // over N*H
    if (idx >= NN * HH) return;
    int i = idx >> 7, j = idx & (HH - 1);
    int p = prev[i];
    float w0j = w0[j], w1j = w0[HH + j], bj = b0[j];
    float xp = x[2 * p] * w0j + x[2 * p + 1] * w1j + bj;
    float xs = x[2 * i] * w0j + x[2 * i + 1] * w1j + bj;
    xin[(size_t)i * K1 + j]      = (_Float16)xp;
    xin[(size_t)i * K1 + HH + j] = (_Float16)xs;
}

// ---- GEMM1: xw[N x 768] = xin[N x 256] @ gat_w[256 x 768] ------------------
// Block: 8 waves = 2(Mw) x 4(Nw); block tile 32M x 256N; A tile via TDM->LDS.
__global__ void __launch_bounds__(256)
gemm_xw_kernel(const _Float16* __restrict__ A, const _Float16* __restrict__ Bp,
               float* __restrict__ C) {
    __shared__ _Float16 shA[32 * K1];                  // 16 KB
    const int lane = threadIdx.x & 31;
    const int wv   = threadIdx.x >> 5;
    const int mb   = blockIdx.x / (NC1 / 256);
    const int nb   = blockIdx.x % (NC1 / 256);
    const int mw   = wv >> 2, nw = wv & 3;

#ifdef HAVE_TDM
    if (threadIdx.x == 0) {
        tdm_load_tile_f16((unsigned int)(uintptr_t)&shA[0],
                          A + (size_t)mb * 32 * K1, K1, NN, 32);
    }
#else
    {   // cooperative fallback: tile is contiguous in memory
        const h8* g = (const h8*)(A + (size_t)mb * 32 * K1);
        h8* s = (h8*)shA;
        for (int i = threadIdx.x; i < 32 * K1 / 8; i += 256) s[i] = g[i];
    }
#endif
    __syncthreads();

    const int lrow = mw * 16 + (lane & 15);            // row within LDS tile
    const int n0   = nb * 256 + nw * 64;
    v8f acc[4] = {};
#pragma unroll
    for (int k0 = 0; k0 < K1; k0 += 32) {
        v16h a = lds_a_frag(shA, K1, lrow, k0, lane);
#pragma unroll
        for (int t = 0; t < 4; ++t) {
            v16h b = packed_b_frag(Bp, NC1 / 16, k0, (n0 >> 4) + t, lane);
            acc[t] = __builtin_amdgcn_wmma_f32_16x16x32_f16(
                false, a, false, b, (short)0, acc[t], false, false);
        }
    }
    const int rowb = mb * 32 + mw * 16 + ((lane & 16) ? 8 : 0);
#pragma unroll
    for (int t = 0; t < 4; ++t) {
        int col = n0 + t * 16 + (lane & 15);
#pragma unroll
        for (int r = 0; r < 8; ++r)
            C[(size_t)(rowb + r) * NC1 + col] = acc[t][r];
    }
}

// ---- attention coefficients: a_s/a_d [N x HEADS] ---------------------------
__global__ void __launch_bounds__(256)
attn_coef_kernel(const float* __restrict__ xw, const float* __restrict__ att_src,
                 const float* __restrict__ att_dst,
                 float* __restrict__ a_s, float* __restrict__ a_d) {
    int idx = blockIdx.x * 256 + threadIdx.x;          // over N*HEADS
    if (idx >= NN * HEADS) return;
    int i = idx / HEADS, h = idx % HEADS;
    const float* row = xw + (size_t)i * NC1 + h * DD;
    const float* ws  = att_src + h * DD;
    const float* wd  = att_dst + h * DD;
    float s = 0.f, d = 0.f;
#pragma unroll 8
    for (int k = 0; k < DD; ++k) { s += row[k] * ws[k]; d += row[k] * wd[k]; }
    a_s[idx] = s;
    a_d[idx] = d;
}

// ---- per-graph dense segment softmax + aggregation + head mean -------------
__global__ void __launch_bounds__(256)
gat_agg_kernel(const float* __restrict__ xw, const float* __restrict__ a_s,
               const float* __restrict__ a_d, const float* __restrict__ gat_b,
               float* __restrict__ enc) {
    __shared__ float asv[HEADS][CC];
    __shared__ float adv[HEADS][CC];
    __shared__ float wsh[CC];
    __shared__ float red[2];

    const int tid   = threadIdx.x;           // output column 0..255
    const int g     = blockIdx.x;
    const int gbase = g * CC;

    if (tid < HEADS * CC) {
        int h = tid / CC, s = tid % CC;
        asv[h][s] = a_s[(gbase + s) * HEADS + h];
        adv[h][s] = a_d[(gbase + s) * HEADS + h];
    }
    __syncthreads();

    const float gb = gat_b[tid];
    for (int d = 0; d < CC; ++d) {
        float acc = 0.f;
        for (int h = 0; h < HEADS; ++h) {
            if (tid < CC) {
                float e = asv[h][tid] + adv[h][d];
                wsh[tid] = (e > 0.f) ? e : 0.2f * e;   // leaky_relu 0.2
            }
            __syncthreads();
            if (tid == 0) {
                float m = wsh[0];
                for (int s = 1; s < CC; ++s) m = fmaxf(m, wsh[s]);
                float den = 0.f;
                for (int s = 0; s < CC; ++s) den += __expf(wsh[s] - m);
                red[0] = m; red[1] = 1.f / den;
            }
            __syncthreads();
            const float m = red[0], inv = red[1];
            const float* xcol = xw + (size_t)gbase * NC1 + h * DD + tid;
#pragma unroll 5
            for (int s = 0; s < CC; ++s) {
                float alpha = __expf(wsh[s] - m) * inv;
                acc += alpha * xcol[(size_t)s * NC1];
            }
            __syncthreads();   // protect wsh/red before next head overwrites
        }
        enc[(size_t)(gbase + d) * DD + tid] = acc * (1.f / HEADS) + gb;
    }
}

// ---- build concat([enc[center], enc]) in f16 -------------------------------
__global__ void __launch_bounds__(256)
prep_cat_kernel(const float* __restrict__ enc, const int* __restrict__ center,
                _Float16* __restrict__ cat) {
    int idx = blockIdx.x * 256 + threadIdx.x;          // over N*D
    if (idx >= NN * DD) return;
    int i = idx >> 8, j = idx & (DD - 1);
    int c = center[i / CC];
    cat[(size_t)i * K2 + j]      = (_Float16)enc[(size_t)c * DD + j];
    cat[(size_t)i * K2 + DD + j] = (_Float16)enc[(size_t)i * DD + j];
}

// ---- GEMM2: h[N x 128] = relu(cat[N x 512] @ w1[512 x 128] + b1) -----------
// Block: 4 waves = 2(Mw) x 2(Nw); block tile 32M x 128N; A tile via TDM->LDS.
__global__ void __launch_bounds__(128)
gemm_h_kernel(const _Float16* __restrict__ A, const _Float16* __restrict__ Bp,
              const float* __restrict__ b1, float* __restrict__ Hout) {
    __shared__ _Float16 shA[32 * K2];                  // 32 KB
    const int lane = threadIdx.x & 31;
    const int wv   = threadIdx.x >> 5;
    const int mb   = blockIdx.x;
    const int mw   = wv >> 1, nw = wv & 1;

#ifdef HAVE_TDM
    if (threadIdx.x == 0) {
        tdm_load_tile_f16((unsigned int)(uintptr_t)&shA[0],
                          A + (size_t)mb * 32 * K2, K2, NN, 32);
    }
#else
    {
        const h8* g = (const h8*)(A + (size_t)mb * 32 * K2);
        h8* s = (h8*)shA;
        for (int i = threadIdx.x; i < 32 * K2 / 8; i += 128) s[i] = g[i];
    }
#endif
    __syncthreads();

    const int lrow = mw * 16 + (lane & 15);
    const int n0   = nw * 64;
    v8f acc[4] = {};
#pragma unroll 4
    for (int k0 = 0; k0 < K2; k0 += 32) {
        v16h a = lds_a_frag(shA, K2, lrow, k0, lane);
#pragma unroll
        for (int t = 0; t < 4; ++t) {
            v16h b = packed_b_frag(Bp, NC2 / 16, k0, (n0 >> 4) + t, lane);
            acc[t] = __builtin_amdgcn_wmma_f32_16x16x32_f16(
                false, a, false, b, (short)0, acc[t], false, false);
        }
    }
    const int rowb = mb * 32 + mw * 16 + ((lane & 16) ? 8 : 0);
#pragma unroll
    for (int t = 0; t < 4; ++t) {
        int col = n0 + t * 16 + (lane & 15);
        float bias = b1[col];
#pragma unroll
        for (int r = 0; r < 8; ++r) {
            float v = acc[t][r] + bias;
            Hout[(size_t)(rowb + r) * NC2 + col] = (v > 0.f) ? v : 0.f;
        }
    }
}

// ---- logits + per-graph masked softmax + greedy sample + log prob ----------
__global__ void __launch_bounds__(64)
head_kernel(const float* __restrict__ x, const float* __restrict__ hbuf,
            const unsigned char* __restrict__ mask, const float* __restrict__ w2,
            const float* __restrict__ b2, const int* __restrict__ graph_id,
            float* __restrict__ out) {
    __shared__ float lg[CC];
    const int g = blockIdx.x, c = threadIdx.x;
    if (c < CC) {
        const int i = g * CC + c;
        float l = b2[0] + x[2 * i] * w2[0] + x[2 * i + 1] * w2[1];
        const float* hr = hbuf + (size_t)i * HH;
#pragma unroll 8
        for (int j = 0; j < HH; ++j) l += hr[j] * w2[2 + j];
        if (!mask[i]) l -= NEG_INF;
        lg[c] = l;
    }
    __syncthreads();
    if (c == 0) {
        float m = lg[0]; int bi = 0;
        for (int s = 1; s < CC; ++s) if (lg[s] > m) { m = lg[s]; bi = s; }
        float den = 0.f;
        for (int s = 0; s < CC; ++s) den += __expf(lg[s] - m);
        out[g]      = (float)(bi + graph_id[g]);       // sample + graph_id_index
        out[GG + g] = (lg[bi] - m) - __logf(den);      // log_action
    }
}

// ---------------------------------------------------------------------------
extern "C" void kernel_launch(void* const* d_in, const int* in_sizes, int n_in,
                              void* d_out, int out_size, void* d_ws, size_t ws_size,
                              hipStream_t stream) {
    const float* x        = (const float*)d_in[0];
    // d_in[1] edge_index: dense complete graph per instance -> not needed
    const int*   prev     = (const int*)d_in[2];
    const int*   center   = (const int*)d_in[3];
    const int*   graph_id = (const int*)d_in[4];
    const unsigned char* mask = (const unsigned char*)d_in[5];
    const float* w0       = (const float*)d_in[6];
    const float* b0       = (const float*)d_in[7];
    const float* gat_w    = (const float*)d_in[8];
    const float* att_src  = (const float*)d_in[9];
    const float* att_dst  = (const float*)d_in[10];
    const float* gat_b    = (const float*)d_in[11];
    const float* w1       = (const float*)d_in[12];
    const float* b1       = (const float*)d_in[13];
    const float* w2       = (const float*)d_in[14];
    const float* b2       = (const float*)d_in[15];
    float* out = (float*)d_out;

    // workspace carve-out (~40 MB)
    char* ws = (char*)d_ws;
    size_t off = 0;
    auto carve = [&](size_t bytes) -> char* {
        char* p = ws + off;
        off += (bytes + 255) & ~(size_t)255;
        return p;
    };
    _Float16* xin   = (_Float16*)carve((size_t)NN * K1 * 2);   // 3.3 MB
    _Float16* gatwp = (_Float16*)carve((size_t)K1 * NC1 * 2);  // 0.4 MB (packed)
    _Float16* w1p   = (_Float16*)carve((size_t)K2 * NC2 * 2);  // 0.13 MB (packed)
    float*    xw    = (float*)   carve((size_t)NN * NC1 * 4);  // 19.7 MB
    float*    a_s   = (float*)   carve((size_t)NN * HEADS * 4);
    float*    a_d   = (float*)   carve((size_t)NN * HEADS * 4);
    float*    enc   = (float*)   carve((size_t)NN * DD * 4);   // 6.6 MB
    _Float16* cat   = (_Float16*)carve((size_t)NN * K2 * 2);   // 6.6 MB
    float*    hbuf  = (float*)   carve((size_t)NN * NC2 * 4);  // 3.3 MB
    (void)ws_size; (void)in_sizes; (void)n_in; (void)out_size;

    // 1. x0 projection + concat(prev, self) -> f16 A matrix
    prep_xin_kernel<<<(NN * HH + 255) / 256, 256, 0, stream>>>(x, prev, w0, b0, xin);

    // 2. pack weights into WMMA B-fragment order (f16)
    pack_b_kernel<<<(K1 * NC1 + 255) / 256, 256, 0, stream>>>(gat_w, gatwp, K1, NC1);
    pack_b_kernel<<<(K2 * NC2 + 255) / 256, 256, 0, stream>>>(w1, w1p, K2, NC2);

    // 3. GEMM1: xw = xin @ gat_w   (200 Mb x 3 Nb blocks, 8 waves each)
    gemm_xw_kernel<<<(NN / 32) * (NC1 / 256), 256, 0, stream>>>(xin, gatwp, xw);

    // 4. attention coefficients
    attn_coef_kernel<<<(NN * HEADS + 255) / 256, 256, 0, stream>>>(xw, att_src, att_dst, a_s, a_d);

    // 5. per-graph softmax + aggregation + head mean + bias
    gat_agg_kernel<<<GG, 256, 0, stream>>>(xw, a_s, a_d, gat_b, enc);

    // 6. concat(center, enc) -> f16
    prep_cat_kernel<<<(NN * DD + 255) / 256, 256, 0, stream>>>(enc, center, cat);

    // 7. GEMM2: h = relu(cat @ w1 + b1)   (200 blocks, 4 waves each)
    gemm_h_kernel<<<NN / 32, 128, 0, stream>>>(cat, w1p, b1, hbuf);

    // 8. logits + masked softmax + argmax + log prob
    head_kernel<<<GG, 64, 0, stream>>>(x, hbuf, mask, w2, b2, graph_id, out);
}
